// LatentVariationalDynamics_24515673326339
// MI455X (gfx1250) — matrix operand order repaired and checked by reading, not compiled, not run on hardware
//
#include <hip/hip_runtime.h>

typedef _Float16 h8  __attribute__((ext_vector_type(8)));
typedef _Float16 h16 __attribute__((ext_vector_type(16)));
typedef float    f8  __attribute__((ext_vector_type(8)));

// global (address_space(1)) pointer types: force global_load/global_store
// instead of flat_* (flat ties up LOADcnt+DScnt and the LDS datapath).
typedef const __attribute__((address_space(1))) _Float16* ghalf;
typedef const __attribute__((address_space(1))) float*    gfloat;
typedef       __attribute__((address_space(1))) float*    gfloatm;
typedef const __attribute__((address_space(1))) h8*       gh8;
#define GCAST(T, p) ((T)(unsigned long long)(p))

#define MT   16      // batch rows per workgroup
#define LDA  264     // LDS activation row stride (halfs), 16B aligned, bank-skewed
#define BSZ  2048
#define TT   8
#define QD   64

enum { EP_TANH = 0, EP_LIN = 1, EP_DERIV = 2, EP_F32 = 3, EP_MASS = 4 };

// Branchless tanh on the hardware transcendental path (v_exp_f32 + v_rcp_f32):
// tanh(x) = sign(x) * (1 - e) / (1 + e),  e = exp(-2|x|); e->0 for large |x|.
__device__ __forceinline__ float tanh_fast(float x) {
  float ax = fabsf(x);
  float e  = __expf(-2.f * ax);
  float t  = (1.f - e) * __builtin_amdgcn_rcpf(1.f + e);
  return copysignf(t, x);
}
// softplus(z) = max(z,0) + log(1 + exp(-|z|)), branchless, exact for large |z|.
__device__ __forceinline__ float softplus_fast(float z) {
  float e = __expf(-fabsf(z));
  return fmaxf(z, 0.f) + __logf(1.f + e);
}

// ---- WMMA fragment loaders (CDNA5 16-bit layouts, wave32) -------------------
// A 16x32 f16 (LDS): lanes 0-15 row M=l, K={0..7,16..23}; lanes 16-31 K={8..15,24..31}
__device__ __forceinline__ h16 load_a_frag(const _Float16* A, int ks, int lane) {
  int row = lane & 15;
  int kb  = ks * 32 + ((lane & 16) ? 8 : 0);
  const _Float16* p = A + row * LDA + kb;
  h8 lo = *(const h8*)(p);
  h8 hi = *(const h8*)(p + 16);
  h16 r;
#pragma unroll
  for (int i = 0; i < 8; ++i) { r[i] = lo[i]; r[i + 8] = hi[i]; }
  return r;
}
// B 32x16 f16 (global): lane holds column N=l&15; lanes 0-15 K={0..7,16..23}, 16-31 K={8..15,24..31}.
// Brow points at row n of an [N][K] (K contiguous) weight image.
__device__ __forceinline__ h16 load_b_frag(ghalf Brow, int ks, int lane) {
  int kb = ks * 32 + ((lane & 16) ? 8 : 0);
  h8 lo = *(gh8)(Brow + kb);
  h8 hi = *(gh8)(Brow + kb + 16);
  h16 r;
#pragma unroll
  for (int i = 0; i < 8; ++i) { r[i] = lo[i]; r[i + 8] = hi[i]; }
  return r;
}

// ---- tiled GEMM: C[16, ntiles*16] = A[16, ksteps*32] x B, fused epilogue ----
template<int EP>
__device__ __forceinline__ void gemm(const _Float16* A, ghalf B, int ldb,
                                     int ksteps, int ntiles, gfloat bias,
                                     const _Float16* act, _Float16* dsth, int ldd,
                                     float* dstf, int ldf, int lane, int wave) {
  for (int nt = wave; nt < ntiles; nt += 8) {      // wave-uniform (EXEC stays all-1s)
    f8 acc = {};
    ghalf Brow = B + (size_t)(nt * 16 + (lane & 15)) * ldb;
    for (int ks = 0; ks < ksteps; ++ks) {
      h16 af = load_a_frag(A, ks, lane);
      h16 bf = load_b_frag(Brow, ks, lane);
      acc = __builtin_amdgcn_wmma_f32_16x16x32_f16(false, af, false, bf,
                                                   (short)0, acc, false, false);
    }
    // C layout: lane col = l&15 (+nt*16); VGPR v -> row v (lanes<16) / v+8
    int col   = nt * 16 + (lane & 15);
    int rbase = (lane & 16) ? 8 : 0;
#pragma unroll
    for (int v = 0; v < 8; ++v) {
      int row = rbase + v;
      float x = acc[v];
      if (EP == EP_TANH) {
        dsth[row * ldd + col] = (_Float16)tanh_fast(x + bias[col]);
      } else if (EP == EP_LIN) {
        dsth[row * ldd + col] = (_Float16)(x + bias[col]);
      } else if (EP == EP_DERIV) {
        float a2 = (float)act[row * LDA + col];
        dsth[row * ldd + col] = (_Float16)(x * (1.f - a2 * a2));
      } else if (EP == EP_F32) {
        dstf[row * ldf + col] = x;
      } else { // EP_MASS: softplus then clamp_min(1e-4)
        dstf[row * ldf + col] = fmaxf(softplus_fast(x + bias[col]), 1e-4f);
      }
    }
  }
}

// ---- gradient of h*L_mlp(x) wrt x for the Lagrangian MLP --------------------
// x = [0.5(qa+qb), (qb-qa)*ih, eta, 0pad]  (K padded 144->160)
// result gx[16][160] f32 (aliased over sAct[1..2])
__device__ void lag_grad(const float* qa, const float* qb, float h, float ih,
                         _Float16* sX, _Float16 (*sAct)[MT * LDA], float* gx,
                         const _Float16* sEta,
                         ghalf lWt1, ghalf lWt2, ghalf lWt3,
                         ghalf lW1p, ghalf lW2n, ghalf lW3n,
                         gfloat lb0, gfloat lb1, gfloat lb2,
                         gfloat lw4, int tid, int lane, int wave) {
  __syncthreads();
  for (int i = tid; i < MT * 160; i += 256) {
    int r = i / 160, c = i - r * 160;
    float v;
    if      (c < 64)  v = 0.5f * (qa[r * 64 + c] + qb[r * 64 + c]);
    else if (c < 128) v = (qb[r * 64 + (c - 64)] - qa[r * 64 + (c - 64)]) * ih;
    else if (c < 144) v = (float)sEta[r * 16 + (c - 128)];
    else              v = 0.f;
    sX[r * LDA + c] = (_Float16)v;
  }
  __syncthreads();
  gemm<EP_TANH>(sX,      lWt1, 160, 5, 16, lb0, nullptr, sAct[0], LDA, nullptr, 0, lane, wave);
  __syncthreads();
  gemm<EP_TANH>(sAct[0], lWt2, 256, 8, 16, lb1, nullptr, sAct[1], LDA, nullptr, 0, lane, wave);
  __syncthreads();
  gemm<EP_TANH>(sAct[1], lWt3, 256, 8, 16, lb2, nullptr, sAct[2], LDA, nullptr, 0, lane, wave);
  __syncthreads();
  // seed: g3 = h * W4[:,0] * (1 - a3^2)  -> sX
  for (int i = tid; i < MT * 256; i += 256) {
    int r = i >> 8, n = i & 255;
    float a3 = (float)sAct[2][r * LDA + n];
    sX[r * LDA + n] = (_Float16)(h * lw4[n] * (1.f - a3 * a3));
  }
  __syncthreads();
  // g2 = (g3 @ W3^T) * (1 - a2^2)  -> sAct[2]   (a3 dead)
  gemm<EP_DERIV>(sX,      lW3n, 256, 8, 16, nullptr, sAct[1], sAct[2], LDA, nullptr, 0, lane, wave);
  __syncthreads();
  // g1 = (g2 @ W2^T) * (1 - a1^2)  -> sX        (g3 dead)
  gemm<EP_DERIV>(sAct[2], lW2n, 256, 8, 16, nullptr, sAct[0], sX, LDA, nullptr, 0, lane, wave);
  __syncthreads();
  // gx = g1 @ W1^T   (N padded to 160)          (a1,a2,g2 dead -> gx aliases them)
  gemm<EP_F32>(sX, lW1p, 256, 8, 10, nullptr, nullptr, nullptr, 0, gx, 160, lane, wave);
  __syncthreads();
}

struct KArgs {
  const float *q0, *q1, *ts;
  const float *cb0, *cb1, *cb2, *cb3;
  const float *lb0, *lb1, *lb2, *lw4;
  const float *mb0, *mb1, *mb2, *mb3;
  const _Float16* ws;
  float* outq;  // [B][T][64]
  float* res;   // scalar accumulator
};

__global__ __launch_bounds__(256) void lvd_kernel(KArgs a) {
  __shared__ _Float16 sX[MT * LDA];
  __shared__ _Float16 sAct[3][MT * LDA];
  __shared__ float    sQ[3][MT * QD];
  __shared__ float    sDL[MT * QD];
  __shared__ float    sMass[MT * QD];
  __shared__ _Float16 sEta[MT * 16];
  __shared__ float    sRow[MT];

  const int tid  = threadIdx.x;
  const int lane = tid & 31, wave = tid >> 5;
  const int row0 = blockIdx.x * MT;
  float* gx = (float*)&sAct[1][0];   // [16][160] f32 over sAct[1..2]

  // global-address-space aliases (emit global_*, not flat_*)
  gfloat  q0g = GCAST(gfloat, a.q0), q1g = GCAST(gfloat, a.q1), tsg = GCAST(gfloat, a.ts);
  gfloat  cb0 = GCAST(gfloat, a.cb0), cb1 = GCAST(gfloat, a.cb1);
  gfloat  cb2 = GCAST(gfloat, a.cb2), cb3 = GCAST(gfloat, a.cb3);
  gfloat  lb0 = GCAST(gfloat, a.lb0), lb1 = GCAST(gfloat, a.lb1);
  gfloat  lb2 = GCAST(gfloat, a.lb2), lw4 = GCAST(gfloat, a.lw4);
  gfloat  mb0 = GCAST(gfloat, a.mb0), mb1 = GCAST(gfloat, a.mb1);
  gfloat  mb2 = GCAST(gfloat, a.mb2), mb3 = GCAST(gfloat, a.mb3);
  gfloatm outg = GCAST(gfloatm, a.outq);

  ghalf cWt1 = GCAST(ghalf, a.ws);
  ghalf cWt2 = cWt1 + 256 * 192;
  ghalf cWt3 = cWt2 + 256 * 256;
  ghalf cWt4 = cWt3 + 256 * 256;
  ghalf lWt1 = cWt4 + 16  * 256;
  ghalf lWt2 = lWt1 + 256 * 160;
  ghalf lWt3 = lWt2 + 256 * 256;
  ghalf lW1p = lWt3 + 256 * 256;
  ghalf lW2n = lW1p + 160 * 256;
  ghalf lW3n = lW2n + 256 * 256;
  ghalf mWt1 = lW3n + 256 * 256;
  ghalf mWt2 = mWt1 + 256 * 96;
  ghalf mWt3 = mWt2 + 256 * 256;
  ghalf mWt4 = mWt3 + 256 * 256;

  // load q0,q1; emit t=0,1
  for (int i = tid; i < MT * QD; i += 256) {
    int r = i >> 6, c = i & 63, b = row0 + r;
    float v0 = q0g[b * QD + c], v1 = q1g[b * QD + c];
    sQ[0][i] = v0; sQ[1][i] = v1;
    outg[(b * TT + 0) * QD + c] = v0;
    outg[(b * TT + 1) * QD + c] = v1;
  }
  __syncthreads();

  // context MLP: eta = MLP([q0,q1,q1-q0])
  for (int i = tid; i < MT * 192; i += 256) {
    int r = i / 192, c = i - r * 192;
    float v;
    if      (c < 64)  v = sQ[0][r * 64 + c];
    else if (c < 128) v = sQ[1][r * 64 + (c - 64)];
    else              v = sQ[1][r * 64 + (c - 128)] - sQ[0][r * 64 + (c - 128)];
    sX[r * LDA + c] = (_Float16)v;
  }
  __syncthreads();
  gemm<EP_TANH>(sX,      cWt1, 192, 6, 16, cb0, nullptr, sAct[0], LDA, nullptr, 0, lane, wave);
  __syncthreads();
  gemm<EP_TANH>(sAct[0], cWt2, 256, 8, 16, cb1, nullptr, sAct[1], LDA, nullptr, 0, lane, wave);
  __syncthreads();
  gemm<EP_TANH>(sAct[1], cWt3, 256, 8, 16, cb2, nullptr, sAct[2], LDA, nullptr, 0, lane, wave);
  __syncthreads();
  gemm<EP_LIN>(sAct[2],  cWt4, 256, 8, 1,  cb3, nullptr, sEta, 16, nullptr, 0, lane, wave);
  __syncthreads();

  float* qp = sQ[0]; float* qc = sQ[1]; float* qn = sQ[2];
  for (int k = 1; k <= TT - 2; ++k) {
    float hp  = fmaxf(fabsf(tsg[k] - tsg[k - 1]), 1e-8f);
    float hn  = fmaxf(fabsf(tsg[k + 1] - tsg[k]), 1e-8f);
    float ihp = 1.f / hp, ihn = 1.f / hn;   // step-uniform scalar divides
    for (int i = tid; i < MT * QD; i += 256) qn[i] = 2.f * qc[i] - qp[i];
    if (tid < MT) sRow[tid] = 0.f;

    // d2_left: grad of h*L at (q_prev, q_curr) -- constant over the 8 iterations
    lag_grad(qp, qc, hp, ihp, sX, sAct, gx, sEta, lWt1, lWt2, lWt3, lW1p, lW2n, lW3n,
             lb0, lb1, lb2, lw4, tid, lane, wave);
    for (int i = tid; i < MT * QD; i += 256) {
      int r = i >> 6, c = i & 63;
      sDL[i] = 0.5f * gx[r * 160 + c] + gx[r * 160 + 64 + c] * ihp;
    }
    __syncthreads();

    // mass = clamp(softplus(MLP([q_curr, eta])), 1e-4) -- constant per step
    for (int i = tid; i < MT * 96; i += 256) {
      int r = i / 96, c = i - r * 96;
      float v;
      if      (c < 64) v = qc[r * 64 + c];
      else if (c < 80) v = (float)sEta[r * 16 + (c - 64)];
      else             v = 0.f;
      sX[r * LDA + c] = (_Float16)v;
    }
    __syncthreads();
    gemm<EP_TANH>(sX,      mWt1,  96, 3, 16, mb0, nullptr, sAct[0], LDA, nullptr, 0, lane, wave);
    __syncthreads();
    gemm<EP_TANH>(sAct[0], mWt2, 256, 8, 16, mb1, nullptr, sAct[1], LDA, nullptr, 0, lane, wave);
    __syncthreads();
    gemm<EP_TANH>(sAct[1], mWt3, 256, 8, 16, mb2, nullptr, sAct[2], LDA, nullptr, 0, lane, wave);
    __syncthreads();
    gemm<EP_MASS>(sAct[2], mWt4, 256, 8, 4,  mb3, nullptr, nullptr, 0, sMass, 64, lane, wave);
    __syncthreads();

    for (int it = 0; it < 8; ++it) {
      lag_grad(qc, qn, hn, ihn, sX, sAct, gx, sEta, lWt1, lWt2, lWt3, lW1p, lW2n, lW3n,
               lb0, lb1, lb2, lw4, tid, lane, wave);
      bool last = (it == 7);
      for (int i = tid; i < MT * QD; i += 256) {
        int r = i >> 6, c = i & 63;
        float d1 = 0.5f * gx[r * 160 + c] - gx[r * 160 + 64 + c] * ihn;
        float rr = sDL[i] + d1;                       // SCALE = 1
        if (isnan(rr)) rr = 0.f;
        rr = fminf(fmaxf(rr, -1e4f), 1e4f);           // nan_to_num + clip
        float upd = 0.05f * rr * __builtin_amdgcn_rcpf(sMass[i]);  // mass >= 1e-4
        upd = fminf(fmaxf(upd, -1.f), 1.f);
        float q = qn[i] - upd;
        if (isnan(q)) q = 0.f;
        else if (isinf(q)) q = (q > 0.f) ? 1e4f : -1e4f;
        qn[i] = q;
        if (last) atomicAdd(&sRow[r], rr * rr);
      }
      __syncthreads();
    }

    if (tid < MT) atomicAdd(a.res, sqrtf(sRow[tid]) * (1.0f / (6.0f * 2048.0f)));
    for (int i = tid; i < MT * QD; i += 256) {
      int r = i >> 6, c = i & 63, b = row0 + r;
      outg[(b * TT + (k + 1)) * QD + c] = qn[i];
    }
    float* t = qp; qp = qc; qc = qn; qn = t;
    __syncthreads();
  }
}

// ---- weight prep: f32 -> f16, transposed ([out][Kpad]) and row-padded -------
__global__ void k_transpose(const float* src, _Float16* dst, int in, int out, int kpad) {
  int idx = blockIdx.x * 256 + threadIdx.x;
  if (idx >= out * kpad) return;
  int o = idx / kpad, i = idx - o * kpad;
  dst[idx] = (i < in) ? (_Float16)src[i * out + o] : (_Float16)0.f;
}
__global__ void k_padrows(const float* src, _Float16* dst, int in, int out, int inpad) {
  int idx = blockIdx.x * 256 + threadIdx.x;
  if (idx >= inpad * out) return;
  int i = idx / out;
  dst[idx] = (i < in) ? (_Float16)src[idx] : (_Float16)0.f;
}
__global__ void k_zero(float* p) { if (blockIdx.x == 0 && threadIdx.x == 0) p[0] = 0.f; }

extern "C" void kernel_launch(void* const* d_in, const int* in_sizes, int n_in,
                              void* d_out, int out_size, void* d_ws, size_t ws_size,
                              hipStream_t stream) {
  (void)in_sizes; (void)n_in; (void)out_size; (void)ws_size;
  const float* q0 = (const float*)d_in[0];
  const float* q1 = (const float*)d_in[1];
  const float* ts = (const float*)d_in[2];
  // pytree flatten order: ctx (W,b)x4, lag (W,b)x4, mass (W,b)x4
  const float* cW0 = (const float*)d_in[3];  const float* cb0 = (const float*)d_in[4];
  const float* cW1 = (const float*)d_in[5];  const float* cb1 = (const float*)d_in[6];
  const float* cW2 = (const float*)d_in[7];  const float* cb2 = (const float*)d_in[8];
  const float* cW3 = (const float*)d_in[9];  const float* cb3 = (const float*)d_in[10];
  const float* lW0 = (const float*)d_in[11]; const float* lb0 = (const float*)d_in[12];
  const float* lW1 = (const float*)d_in[13]; const float* lb1 = (const float*)d_in[14];
  const float* lW2 = (const float*)d_in[15]; const float* lb2 = (const float*)d_in[16];
  const float* lW3 = (const float*)d_in[17];
  const float* mW0 = (const float*)d_in[19]; const float* mb0 = (const float*)d_in[20];
  const float* mW1 = (const float*)d_in[21]; const float* mb1 = (const float*)d_in[22];
  const float* mW2 = (const float*)d_in[23]; const float* mb2 = (const float*)d_in[24];
  const float* mW3 = (const float*)d_in[25]; const float* mb3 = (const float*)d_in[26];

  _Float16* ws   = (_Float16*)d_ws;
  _Float16* cWt1 = ws;
  _Float16* cWt2 = cWt1 + 256 * 192;
  _Float16* cWt3 = cWt2 + 256 * 256;
  _Float16* cWt4 = cWt3 + 256 * 256;
  _Float16* lWt1 = cWt4 + 16  * 256;
  _Float16* lWt2 = lWt1 + 256 * 160;
  _Float16* lWt3 = lWt2 + 256 * 256;
  _Float16* lW1p = lWt3 + 256 * 256;
  _Float16* lW2n = lW1p + 160 * 256;
  _Float16* lW3n = lW2n + 256 * 256;
  _Float16* mWt1 = lW3n + 256 * 256;
  _Float16* mWt2 = mWt1 + 256 * 96;
  _Float16* mWt3 = mWt2 + 256 * 256;
  _Float16* mWt4 = mWt3 + 256 * 256;

#define TLAUNCH(s, d, in_, out_, kp) do { int tot = (out_) * (kp); \
    k_transpose<<<(tot + 255) / 256, 256, 0, stream>>>(s, d, in_, out_, kp); } while (0)
#define PLAUNCH(s, d, in_, out_, ip) do { int tot = (ip) * (out_); \
    k_padrows<<<(tot + 255) / 256, 256, 0, stream>>>(s, d, in_, out_, ip); } while (0)

  TLAUNCH(cW0, cWt1, 192, 256, 192);
  TLAUNCH(cW1, cWt2, 256, 256, 256);
  TLAUNCH(cW2, cWt3, 256, 256, 256);
  TLAUNCH(cW3, cWt4, 256, 16,  256);
  TLAUNCH(lW0, lWt1, 144, 256, 160);
  TLAUNCH(lW1, lWt2, 256, 256, 256);
  TLAUNCH(lW2, lWt3, 256, 256, 256);
  PLAUNCH(lW0, lW1p, 144, 256, 160);
  PLAUNCH(lW1, lW2n, 256, 256, 256);
  PLAUNCH(lW2, lW3n, 256, 256, 256);
  TLAUNCH(mW0, mWt1, 80,  256, 96);
  TLAUNCH(mW1, mWt2, 256, 256, 256);
  TLAUNCH(mW2, mWt3, 256, 256, 256);
  TLAUNCH(mW3, mWt4, 256, 64,  256);

  float* outq = (float*)d_out;
  float* res  = outq + (size_t)BSZ * TT * QD;
  k_zero<<<1, 32, 0, stream>>>(res);

  KArgs args;
  args.q0 = q0; args.q1 = q1; args.ts = ts;
  args.cb0 = cb0; args.cb1 = cb1; args.cb2 = cb2; args.cb3 = cb3;
  args.lb0 = lb0; args.lb1 = lb1; args.lb2 = lb2; args.lw4 = lW3;
  args.mb0 = mb0; args.mb1 = mb1; args.mb2 = mb2; args.mb3 = mb3;
  args.ws = ws; args.outq = outq; args.res = res;
  lvd_kernel<<<BSZ / MT, 256, 0, stream>>>(args);
}
